// MultiHeadAttention_24489903522408
// MI455X (gfx1250) — compile-verified, matrix operand order checked by reading, not run on hardware
//
#include <hip/hip_runtime.h>
#include <hip/hip_bf16.h>
#include <stdint.h>

#define HIDDEN 1024
#define NHEADS 16
#define DK     64
#define SEQ    2048
#define BATCH  4
#define NEGINF (-1e9f)

typedef __attribute__((ext_vector_type(16))) __bf16 v16bf;
typedef __attribute__((ext_vector_type(8)))  float  v8f;

union __align__(16) Frag16 {
    v16bf v;
    uint4 q[2];
    unsigned short s[16];
};

__device__ __forceinline__ unsigned short f2bf(float x) {
    unsigned int u = __float_as_uint(x);
    unsigned int r = u + 0x7FFFu + ((u >> 16) & 1u);   // round-to-nearest-even
    return (unsigned short)(r >> 16);
}

// Load a 16x32 bf16 fragment (A layout; B layout is the mirrored same math)
// from a row-major matrix with leading dim `ld`, starting at (r0, k0).
// Per CDNA5 ISA 16-bit A 16x32 layout:
//   lanes 0-15 : row = r0+lane,    e[0..7]=K[k0..k0+7],   e[8..15]=K[k0+16..k0+23]
//   lanes16-31 : row = r0+lane-16, e[0..7]=K[k0+8..k0+15],e[8..15]=K[k0+24..k0+31]
__device__ __forceinline__ v16bf load_frag(const unsigned short* __restrict__ base,
                                           int ld, int r0, int k0, int lane) {
    const int half = lane >> 4;
    const int r    = r0 + (lane & 15);
    const unsigned short* p = base + (size_t)r * ld + k0 + half * 8;
    Frag16 f;
    f.q[0] = *(const uint4*)(p);
    f.q[1] = *(const uint4*)(p + 16);
    return f.v;
}

__device__ __forceinline__ v8f wmma_bf16(v16bf a, v16bf b, v8f c) {
    return __builtin_amdgcn_wmma_f32_16x16x32_bf16(
        /*neg_a=*/false, a, /*neg_b=*/false, b,
        /*c_mod=*/(short)0, c, /*reuse_a=*/false, /*reuse_b=*/false);
}

// ---------------------------------------------------------------------------
// fp32 -> bf16 conversion (grid-stride)
// ---------------------------------------------------------------------------
__global__ void cvt_f32_bf16(const float* __restrict__ in,
                             unsigned short* __restrict__ out, int n) {
    int i = blockIdx.x * blockDim.x + threadIdx.x;
    int stride = gridDim.x * blockDim.x;
    for (; i < n; i += stride) out[i] = f2bf(in[i]);
}

// ---------------------------------------------------------------------------
// GEMM: C[M,N] = A[M,K](bf16) * W[N,K](bf16)^T + bias[N]
// Block = 256 threads (8 waves), tile 128x128, wave tile 32x64.
// mode 0: scatter bf16 into [B,H,S,DK]      (Q and K projections)
// mode 1: scatter bf16 into [B,H,DK,S]      (V projection, transposed)
// mode 2: fp32 row-major [M,HIDDEN]         (output projection -> d_out)
// ---------------------------------------------------------------------------
__global__ __launch_bounds__(256) void gemm_bf16_128x128(
    const unsigned short* __restrict__ A, const unsigned short* __restrict__ W,
    const float* __restrict__ bias, void* __restrict__ out, int K, int mode) {
    const int lane = threadIdx.x & 31;
    const int wid  = threadIdx.x >> 5;
    const int half = lane >> 4;
    const int col  = lane & 15;
    const int rb = blockIdx.y * 128 + (wid & 3) * 32;   // row base of wave tile
    const int cb = blockIdx.x * 128 + (wid >> 2) * 64;  // col base of wave tile

    v8f acc[2][4];
#pragma unroll
    for (int r = 0; r < 2; ++r)
#pragma unroll
        for (int t = 0; t < 4; ++t) acc[r][t] = (v8f)0.0f;

    for (int kk = 0; kk < K; kk += 32) {
        v16bf a0 = load_frag(A, K, rb,      kk, lane);
        v16bf a1 = load_frag(A, K, rb + 16, kk, lane);
#pragma unroll
        for (int t = 0; t < 4; ++t) {
            v16bf b = load_frag(W, K, cb + t * 16, kk, lane);
            acc[0][t] = wmma_bf16(a0, b, acc[0][t]);
            acc[1][t] = wmma_bf16(a1, b, acc[1][t]);
        }
    }

    // Epilogue. C layout: VGPR v, row = v + 8*(lane>>4), col = lane&15 (per tile).
#pragma unroll
    for (int r = 0; r < 2; ++r) {
#pragma unroll
        for (int t = 0; t < 4; ++t) {
            const int n  = cb + t * 16 + col;
            const float bv = bias[n];
#pragma unroll
            for (int v = 0; v < 8; ++v) {
                const int m = rb + r * 16 + v + 8 * half;
                const float val = acc[r][t][v] + bv;
                if (mode == 2) {
                    ((float*)out)[(size_t)m * HIDDEN + n] = val;
                } else {
                    const int bb = m >> 11, s = m & (SEQ - 1);
                    const int hh = n >> 6,  d = n & 63;
                    const size_t idx = (mode == 0)
                        ? (((size_t)(bb * NHEADS + hh) * SEQ + s) * DK + d)
                        : (((size_t)(bb * NHEADS + hh) * DK + d) * SEQ + s);
                    ((unsigned short*)out)[idx] = f2bf(val);
                }
            }
        }
    }
}

// ---------------------------------------------------------------------------
// Flash-style attention.
//   Qp, Kp : [B,H,S,DK] bf16     Vt : [B,H,DK,S] bf16
//   Cx     : [B,S,HIDDEN] bf16   (context, ready for output projection)
// Block = 256 threads (8 waves); wave owns 16 query rows; 128 q rows / block.
// ---------------------------------------------------------------------------
__global__ __launch_bounds__(256) void attn_bf16(
    const unsigned short* __restrict__ Qp, const unsigned short* __restrict__ Kp,
    const unsigned short* __restrict__ Vt, const unsigned char* __restrict__ mask,
    unsigned short* __restrict__ Cx) {
    __shared__ __align__(16) unsigned short pbuf[8][16 * 32];  // per-wave P tile

    const int lane = threadIdx.x & 31;
    const int wid  = threadIdx.x >> 5;
    const int half = lane >> 4;
    const int col  = lane & 15;
    const int bh = blockIdx.y;
    const int b  = bh >> 4, h = bh & 15;
    const int q0 = blockIdx.x * 128 + wid * 16;

    const unsigned short* Qb = Qp + (size_t)bh * SEQ * DK;
    const unsigned short* Kb = Kp + (size_t)bh * SEQ * DK;
    const unsigned short* Vb = Vt + (size_t)bh * DK * SEQ;
    const unsigned char*  mb = mask + (size_t)b * SEQ;

    // Q fragments (resident): 16 rows x 64 d, two 32-wide K-steps.
    v16bf qf0 = load_frag(Qb, DK, q0, 0,  lane);
    v16bf qf1 = load_frag(Qb, DK, q0, 32, lane);

    v8f ctx[4];
#pragma unroll
    for (int t = 0; t < 4; ++t) ctx[t] = (v8f)0.0f;
    float rmax[8], rsum[8];
#pragma unroll
    for (int v = 0; v < 8; ++v) { rmax[v] = -3.4e38f; rsum[v] = 0.0f; }

    for (int kb = 0; kb < SEQ; kb += 32) {
        // scores = Q (16x64) x K^T (64x32) / sqrt(dk)
        v8f sc[2];
        sc[0] = (v8f)0.0f; sc[1] = (v8f)0.0f;
#pragma unroll
        for (int tc = 0; tc < 2; ++tc) {
            v16bf kfa = load_frag(Kb, DK, kb + tc * 16, 0,  lane);
            sc[tc] = wmma_bf16(qf0, kfa, sc[tc]);
            v16bf kfb = load_frag(Kb, DK, kb + tc * 16, 32, lane);
            sc[tc] = wmma_bf16(qf1, kfb, sc[tc]);
        }
        const float madd0 = mb[kb + col]      ? 0.0f : NEGINF;
        const float madd1 = mb[kb + 16 + col] ? 0.0f : NEGINF;
#pragma unroll
        for (int v = 0; v < 8; ++v) {
            sc[0][v] = sc[0][v] * 0.125f + madd0;   // 1/sqrt(64)
            sc[1][v] = sc[1][v] * 0.125f + madd1;
        }

        // online softmax: rows live in a fixed 16-lane group (same lane>>4);
        // xor masks 1,2,4,8 reduce across columns only.
        float p0[8], p1[8];
#pragma unroll
        for (int v = 0; v < 8; ++v) {
            float t = fmaxf(sc[0][v], sc[1][v]);
            t = fmaxf(t, __shfl_xor(t, 1, 32));
            t = fmaxf(t, __shfl_xor(t, 2, 32));
            t = fmaxf(t, __shfl_xor(t, 4, 32));
            t = fmaxf(t, __shfl_xor(t, 8, 32));
            const float nm = fmaxf(rmax[v], t);
            const float f  = __expf(rmax[v] - nm);
            rmax[v] = nm;
            p0[v] = __expf(sc[0][v] - nm);
            p1[v] = __expf(sc[1][v] - nm);
            float rs = p0[v] + p1[v];
            rs += __shfl_xor(rs, 1, 32);
            rs += __shfl_xor(rs, 2, 32);
            rs += __shfl_xor(rs, 4, 32);
            rs += __shfl_xor(rs, 8, 32);
            rsum[v] = rsum[v] * f + rs;
#pragma unroll
            for (int t2 = 0; t2 < 4; ++t2) ctx[t2][v] *= f;
        }

        // C-layout -> A-layout via per-wave LDS round trip.
        unsigned short* pl = pbuf[wid];
#pragma unroll
        for (int v = 0; v < 8; ++v) {
            const int m = v + 8 * half;
            pl[m * 32 + col]      = f2bf(p0[v]);
            pl[m * 32 + 16 + col] = f2bf(p1[v]);
        }
        asm volatile("s_wait_dscnt 0" ::: "memory");  // wave-local DS RAW fence
        Frag16 pf;
        pf.q[0] = *(const uint4*)(pl + col * 32 + half * 8);
        pf.q[1] = *(const uint4*)(pl + col * 32 + 16 + half * 8);

        // ctx (16x64) += P (16x32) x V (32x64); V stored transposed so the
        // B fragment is a contiguous-K load: B[k,n] = Vt[n, kb+k].
#pragma unroll
        for (int t2 = 0; t2 < 4; ++t2) {
            v16bf vf = load_frag(Vb, SEQ, t2 * 16, kb, lane);
            ctx[t2] = wmma_bf16(pf.v, vf, ctx[t2]);
        }
    }

    // normalize and write context into [B, S, HIDDEN] bf16
#pragma unroll
    for (int v = 0; v < 8; ++v) {
        const float inv = 1.0f / rsum[v];
        const int qr = q0 + v + 8 * half;
        const size_t rowbase = ((size_t)b * SEQ + qr) * HIDDEN + h * DK;
#pragma unroll
        for (int t2 = 0; t2 < 4; ++t2)
            Cx[rowbase + t2 * 16 + col] = f2bf(ctx[t2][v] * inv);
    }
}

// ---------------------------------------------------------------------------
// launcher
// ---------------------------------------------------------------------------
extern "C" void kernel_launch(void* const* d_in, const int* in_sizes, int n_in,
                              void* d_out, int out_size, void* d_ws, size_t ws_size,
                              hipStream_t stream) {
    const float* query = (const float*)d_in[0];
    const float* key_  = (const float*)d_in[1];
    const float* value = (const float*)d_in[2];
    const unsigned char* mask = (const unsigned char*)d_in[3];
    const float* w_q = (const float*)d_in[4];
    const float* b_q = (const float*)d_in[5];
    const float* w_k = (const float*)d_in[6];
    const float* b_k = (const float*)d_in[7];
    const float* w_v = (const float*)d_in[8];
    const float* b_v = (const float*)d_in[9];
    const float* w_o = (const float*)d_in[10];
    const float* b_o = (const float*)d_in[11];

    const size_t ACT = (size_t)BATCH * SEQ * HIDDEN;   // 8,388,608 elements
    const size_t WEL = (size_t)HIDDEN * HIDDEN;        // 1,048,576 elements

    unsigned short* qx  = (unsigned short*)d_ws;
    unsigned short* kx  = qx  + ACT;
    unsigned short* vx  = kx  + ACT;
    unsigned short* wqb = vx  + ACT;
    unsigned short* wkb = wqb + WEL;
    unsigned short* wvb = wkb + WEL;
    unsigned short* wob = wvb + WEL;
    unsigned short* Qp  = wob + WEL;    // [B,H,S,DK]
    unsigned short* Kp  = Qp  + ACT;    // [B,H,S,DK]
    unsigned short* Vt  = Kp  + ACT;    // [B,H,DK,S]
    unsigned short* Cx  = Vt  + ACT;    // [B,S,HIDDEN]

    // 1) fp32 -> bf16
    cvt_f32_bf16<<<512, 256, 0, stream>>>(query, qx, (int)ACT);
    cvt_f32_bf16<<<512, 256, 0, stream>>>(key_,  kx, (int)ACT);
    cvt_f32_bf16<<<512, 256, 0, stream>>>(value, vx, (int)ACT);
    cvt_f32_bf16<<<256, 256, 0, stream>>>(w_q, wqb, (int)WEL);
    cvt_f32_bf16<<<256, 256, 0, stream>>>(w_k, wkb, (int)WEL);
    cvt_f32_bf16<<<256, 256, 0, stream>>>(w_v, wvb, (int)WEL);
    cvt_f32_bf16<<<256, 256, 0, stream>>>(w_o, wob, (int)WEL);

    // 2) projections (M = B*S = 8192, N = K = 1024)
    dim3 gblk(256);
    dim3 ggrid(HIDDEN / 128, (BATCH * SEQ) / 128);     // (8, 64)
    gemm_bf16_128x128<<<ggrid, gblk, 0, stream>>>(qx, wqb, b_q, Qp, HIDDEN, 0);
    gemm_bf16_128x128<<<ggrid, gblk, 0, stream>>>(kx, wkb, b_k, Kp, HIDDEN, 0);
    gemm_bf16_128x128<<<ggrid, gblk, 0, stream>>>(vx, wvb, b_v, Vt, HIDDEN, 1);

    // 3) attention (grid: 2048/128 q-blocks x B*H)
    attn_bf16<<<dim3(SEQ / 128, BATCH * NHEADS), 256, 0, stream>>>(Qp, Kp, Vt, mask, Cx);

    // 4) output projection -> fp32 d_out
    gemm_bf16_128x128<<<ggrid, gblk, 0, stream>>>(Cx, wob, b_o, d_out, HIDDEN, 2);
}
